// AttentionEncoder_71219147702374
// MI455X (gfx1250) — matrix operand order repaired and checked by reading, not compiled, not run on hardware
//
#include <hip/hip_runtime.h>
#include <stddef.h>

// ---------------------------------------------------------------------------
// Problem constants (match reference)
// ---------------------------------------------------------------------------
#define BB   64
#define TT   256
#define LL   256
#define DIN  1024
#define DC   1024
#define HH   1024
#define AA   1024
#define GG   (3 * HH)   // 3072

typedef __attribute__((ext_vector_type(16))) __bf16 bfrag16;
typedef __attribute__((ext_vector_type(8)))  __bf16 bvec8;
typedef __attribute__((ext_vector_type(8)))  float  v8f;

// ---------------------------------------------------------------------------
// f32 -> bf16 flat conversion (n multiple of 256)
// ---------------------------------------------------------------------------
__global__ __launch_bounds__(256)
void cvt_bf16_kernel(const float* __restrict__ in, __bf16* __restrict__ out)
{
    const size_t i = (size_t)blockIdx.x * blockDim.x + threadIdx.x;
    out[i] = (__bf16)in[i];
}

__global__ __launch_bounds__(256)
void zero_u32_kernel(unsigned int* __restrict__ p)
{
    p[(size_t)blockIdx.x * blockDim.x + threadIdx.x] = 0u;
}

// ---------------------------------------------------------------------------
// Pack a row-major f32 weight matrix (K x N, row stride ldw) into bf16
// wave32 WMMA B-fragment order:
//   fragment f = kt*(N/16)+nt ; within fragment: 32 lanes x 16 bf16 (contig).
//   lane<16 : n = nt*16+lane,    k = kt*32 + (e<8 ? e : 16+e-8)      (e = 0..15)
//   lane>=16: n = nt*16+lane-16, k = kt*32 + 8 + (e<8 ? e : 16+e-8)
// ---------------------------------------------------------------------------
__global__ __launch_bounds__(256)
void pack_b_kernel(const float* __restrict__ W, long ldw,
                   __bf16* __restrict__ out, int K, int N)
{
    int idx = blockIdx.x * blockDim.x + threadIdx.x;
    if (idx >= K * N) return;
    int e  = idx & 15;
    int l  = (idx >> 4) & 31;
    int f  = idx >> 9;
    int Ntiles = N >> 4;
    int nt = f % Ntiles;
    int kt = f / Ntiles;
    int k  = kt * 32 + ((e >> 3) << 4) + ((l >> 4) << 3) + (e & 7);
    int n  = nt * 16 + (l & 15);
    out[idx] = (__bf16)W[(size_t)k * ldw + n];
}

// ---------------------------------------------------------------------------
// WMMA bf16 GEMM:  C(MxN) = A(MxK bf16 row-major, row stride lda) * Bpacked
// Templated epilogue: optional bias[n], accumulate into Cf, or bf16 output.
// Block = 128 threads (4 waves), tile 64x64; wave w -> rows [w*16, w*16+16).
// Software-pipelined: kt+1 fragments loaded before kt's WMMAs issue.
// ---------------------------------------------------------------------------
template <bool HAS_BIAS, bool ACCUM, bool BF16_OUT>
__global__ __launch_bounds__(128)
void wmma_gemm(const __bf16* __restrict__ A, long lda,
               const __bf16* __restrict__ Bp,
               const float* __restrict__ bias,
               float* __restrict__ Cf,
               __bf16* __restrict__ Cbf,
               int M, int N, int K)
{
    const int tid  = threadIdx.x;
    const int wave = tid >> 5;
    const int lane = tid & 31;
    const int rowBase = blockIdx.y * 64 + wave * 16;
    const int colBase = blockIdx.x * 64;
    const int Ntiles  = N >> 4;
    const int Kt      = K >> 5;

    const bfrag16* __restrict__ Bv = reinterpret_cast<const bfrag16*>(Bp);

    const int mrow  = rowBase + (lane & 15);
    const int khalf = lane >> 4;
    const __bf16* __restrict__ Arow = A + (size_t)mrow * lda + khalf * 8;
    const int nt0 = colBase >> 4;

    v8f acc0 = {}, acc1 = {}, acc2 = {}, acc3 = {};

    auto loadA = [&](int kt) -> bfrag16 {
        bvec8 lo = *reinterpret_cast<const bvec8*>(Arow + kt * 32);
        bvec8 hi = *reinterpret_cast<const bvec8*>(Arow + kt * 32 + 16);
        return __builtin_shufflevector(lo, hi, 0, 1, 2, 3, 4, 5, 6, 7,
                                               8, 9, 10, 11, 12, 13, 14, 15);
    };
    auto loadB = [&](int kt, int j) -> bfrag16 {
        return Bv[((size_t)kt * Ntiles + nt0 + j) * 32 + lane];
    };

    bfrag16 a_cur = loadA(0);
    bfrag16 b0 = loadB(0, 0), b1 = loadB(0, 1), b2 = loadB(0, 2), b3 = loadB(0, 3);

    for (int kt = 0; kt < Kt - 1; ++kt) {
        // Issue next iteration's loads first so they overlap the WMMAs.
        bfrag16 a_nxt = loadA(kt + 1);
        bfrag16 n0 = loadB(kt + 1, 0);
        bfrag16 n1 = loadB(kt + 1, 1);
        bfrag16 n2 = loadB(kt + 1, 2);
        bfrag16 n3 = loadB(kt + 1, 3);
        if (kt + 2 < Kt)
            __builtin_prefetch(&Bv[((size_t)(kt + 2) * Ntiles + nt0) * 32 + lane], 0, 1);

        acc0 = __builtin_amdgcn_wmma_f32_16x16x32_bf16(false, a_cur, false, b0, (short)0, acc0, false, false);
        acc1 = __builtin_amdgcn_wmma_f32_16x16x32_bf16(false, a_cur, false, b1, (short)0, acc1, false, false);
        acc2 = __builtin_amdgcn_wmma_f32_16x16x32_bf16(false, a_cur, false, b2, (short)0, acc2, false, false);
        acc3 = __builtin_amdgcn_wmma_f32_16x16x32_bf16(false, a_cur, false, b3, (short)0, acc3, false, false);

        a_cur = a_nxt; b0 = n0; b1 = n1; b2 = n2; b3 = n3;
    }
    // Peeled last iteration.
    acc0 = __builtin_amdgcn_wmma_f32_16x16x32_bf16(false, a_cur, false, b0, (short)0, acc0, false, false);
    acc1 = __builtin_amdgcn_wmma_f32_16x16x32_bf16(false, a_cur, false, b1, (short)0, acc1, false, false);
    acc2 = __builtin_amdgcn_wmma_f32_16x16x32_bf16(false, a_cur, false, b2, (short)0, acc2, false, false);
    acc3 = __builtin_amdgcn_wmma_f32_16x16x32_bf16(false, a_cur, false, b3, (short)0, acc3, false, false);

    // Epilogue. C/D layout: VGPR i -> row = rowBase + i + 8*(lane>>4),
    //                       col = colBase + j*16 + (lane&15)
    const int mro = rowBase + 8 * khalf;
    const int nco = lane & 15;
    v8f accs[4] = {acc0, acc1, acc2, acc3};
#pragma unroll
    for (int j = 0; j < 4; ++j) {
        const int n = colBase + j * 16 + nco;
        float bv = 0.0f;
        if (HAS_BIAS) bv = bias[n];
#pragma unroll
        for (int i = 0; i < 8; ++i) {
            const int m = mro + i;
            const size_t o = (size_t)m * N + n;
            float val = accs[j][i] + bv;
            if (BF16_OUT) {
                Cbf[o] = (__bf16)val;
            } else if (ACCUM) {
                Cf[o] += val;
            } else {
                Cf[o] = val;
            }
        }
    }
}

// ---------------------------------------------------------------------------
// Fused additive attention for one timestep, one block per batch row b:
//   e[l]  = sum_a v[a] * tanh(ctxp[b,l,a] + hWh[b,a])
//   a[l]  = softmax_l(e)
//   c[d]  = sum_l a[l] * context[b,l,d]      (emitted directly as bf16)
// ---------------------------------------------------------------------------
__global__ __launch_bounds__(256)
void attn_kernel(const __bf16* __restrict__ ctxp,     // (B,L,A) bf16
                 const float*  __restrict__ hWh,      // (B,A)
                 const float*  __restrict__ vvec,     // (A)
                 const float*  __restrict__ context,  // (B,L,DC) f32
                 __bf16*       __restrict__ cbf)      // (B,DC) bf16
{
    const int b   = blockIdx.x;
    const int tid = threadIdx.x;

    __shared__ float s_h[AA];
    __shared__ float s_v[AA];
    __shared__ float s_e[LL];
    __shared__ float s_red[256];

    for (int a = tid; a < AA; a += 256) {
        s_h[a] = hWh[(size_t)b * AA + a];
        s_v[a] = vvec[a];
    }
    __syncthreads();

    const int warp = tid >> 5;
    const int lane = tid & 31;
    for (int l = warp; l < LL; l += 8) {
        const bvec8* row = reinterpret_cast<const bvec8*>(
            ctxp + ((size_t)b * LL + l) * AA);
        float acc = 0.0f;
        for (int c = lane; c < AA / 8; c += 32) {     // coalesced 16B per lane
            bvec8 x = row[c];
            const int a0 = c * 8;
#pragma unroll
            for (int e = 0; e < 8; ++e)
                acc += s_v[a0 + e] * tanhf((float)x[e] + s_h[a0 + e]);
        }
#pragma unroll
        for (int off = 16; off > 0; off >>= 1)
            acc += __shfl_down(acc, off, 32);
        if (lane == 0) s_e[l] = acc;
    }
    __syncthreads();

    // softmax over L=256 (one element per thread)
    const float x = s_e[tid];
    s_red[tid] = x;
    __syncthreads();
    for (int s = 128; s > 0; s >>= 1) {
        if (tid < s) s_red[tid] = fmaxf(s_red[tid], s_red[tid + s]);
        __syncthreads();
    }
    const float mx = s_red[0];
    __syncthreads();
    const float ex = __expf(x - mx);
    s_red[tid] = ex;
    __syncthreads();
    for (int s = 128; s > 0; s >>= 1) {
        if (tid < s) s_red[tid] += s_red[tid + s];
        __syncthreads();
    }
    const float inv = 1.0f / s_red[0];
    __syncthreads();
    s_e[tid] = ex * inv;
    __syncthreads();

    // weighted context sum
    for (int d = tid; d < DC; d += 256) {
        float acc = 0.0f;
        const float* cb = context + (size_t)b * LL * DC + d;
#pragma unroll 4
        for (int l = 0; l < LL; ++l)
            acc += s_e[l] * cb[(size_t)l * DC];
        cbf[(size_t)b * DC + d] = (__bf16)acc;
    }
}

// ---------------------------------------------------------------------------
// GRU gate fusion; updates h (f32 + bf16 mirror), writes output slice for t.
// ---------------------------------------------------------------------------
__global__ __launch_bounds__(256)
void gate_kernel(const float* __restrict__ gi, const float* __restrict__ gh,
                 float* __restrict__ h, __bf16* __restrict__ hbf,
                 float* __restrict__ out, int t, int dirOff)
{
    const int idx = blockIdx.x * blockDim.x + threadIdx.x;  // B*H
    const int b = idx >> 10;
    const int j = idx & 1023;
    const size_t gb = (size_t)b * GG;
    const float ir = gi[gb + j], iz = gi[gb + HH + j], in = gi[gb + 2 * HH + j];
    const float hr = gh[gb + j], hz = gh[gb + HH + j], hn = gh[gb + 2 * HH + j];
    const float r = 1.0f / (1.0f + __expf(-(ir + hr)));
    const float u = 1.0f / (1.0f + __expf(-(iz + hz)));
    const float n = tanhf(in + r * hn);
    const float hp = h[idx];
    const float hnew = (1.0f - u) * n + u * hp;
    h[idx]   = hnew;
    hbf[idx] = (__bf16)hnew;
    out[((size_t)t * BB + b) * (2 * HH) + dirOff + j] = hnew;
}

__global__ __launch_bounds__(256)
void hidden_kernel(const float* __restrict__ hf, const float* __restrict__ hr,
                   float* __restrict__ hid)
{
    const int idx = blockIdx.x * blockDim.x + threadIdx.x;  // B*H
    const int b = idx >> 10;
    const int j = idx & 1023;
    hid[(size_t)b * (2 * HH) + j]      = hf[idx];
    hid[(size_t)b * (2 * HH) + HH + j] = hr[idx];
}

// ---------------------------------------------------------------------------
// Host launcher
// ---------------------------------------------------------------------------
extern "C" void kernel_launch(void* const* d_in, const int* in_sizes, int n_in,
                              void* d_out, int out_size, void* d_ws, size_t ws_size,
                              hipStream_t stream)
{
    (void)in_sizes; (void)n_in; (void)out_size; (void)ws_size;

    const float* query   = (const float*)d_in[0];   // (B,T,DIN)
    const float* context = (const float*)d_in[1];   // (B,L,DC)
    struct DirW { const float *Wc, *Wh, *v, *Wih, *bih, *Whh, *bhh; };
    DirW dw[2];
    for (int d = 0; d < 2; ++d) {
        const int o = 2 + 7 * d;
        dw[d].Wc  = (const float*)d_in[o + 0];
        dw[d].Wh  = (const float*)d_in[o + 1];
        dw[d].v   = (const float*)d_in[o + 2];
        dw[d].Wih = (const float*)d_in[o + 3];
        dw[d].bih = (const float*)d_in[o + 4];
        dw[d].Whh = (const float*)d_in[o + 5];
        dw[d].bhh = (const float*)d_in[o + 6];
    }
    float* out    = (float*)d_out;
    float* hidden = out + (size_t)TT * BB * 2 * HH;

    // ---- carve workspace (256B aligned regions) ----
    char* ws = (char*)d_ws;
    size_t off = 0;
    auto carve = [&](size_t bytes) -> void* {
        void* p = ws + off;
        off += (bytes + 255) & ~(size_t)255;
        return p;
    };
    __bf16 *Wc_p[2], *Wh_p[2], *WihT_p[2], *WihB_p[2], *Whh_p[2];
    for (int d = 0; d < 2; ++d) {
        Wc_p[d]   = (__bf16*)carve((size_t)DC  * AA * 2);
        Wh_p[d]   = (__bf16*)carve((size_t)HH  * AA * 2);
        WihT_p[d] = (__bf16*)carve((size_t)DIN * GG * 2);
        WihB_p[d] = (__bf16*)carve((size_t)DC  * GG * 2);
        Whh_p[d]  = (__bf16*)carve((size_t)HH  * GG * 2);
    }
    __bf16* ctxp   = (__bf16*)carve((size_t)BB * LL * AA * 2);  // reused per dir
    __bf16* q_bf   = (__bf16*)carve((size_t)BB * TT * DIN * 2);
    __bf16* ctx_bf = (__bf16*)carve((size_t)BB * LL * DC * 2);
    __bf16* c_bf   = (__bf16*)carve((size_t)BB * DC * 2);
    float* hWh  = (float*)carve((size_t)BB * AA * 4);
    float* gi   = (float*)carve((size_t)BB * GG * 4);
    float* gh   = (float*)carve((size_t)BB * GG * 4);
    float*  hbuf[2];
    __bf16* hbf[2];
    hbuf[0] = (float*)carve((size_t)BB * HH * 4);
    hbuf[1] = (float*)carve((size_t)BB * HH * 4);
    hbf[0]  = (__bf16*)carve((size_t)BB * HH * 2);
    hbf[1]  = (__bf16*)carve((size_t)BB * HH * 2);

    // ---- one-time: pack weights + convert activations to bf16 ----
    cvt_bf16_kernel<<<(BB * TT * DIN) / 256, 256, 0, stream>>>(query, q_bf);
    cvt_bf16_kernel<<<(BB * LL * DC) / 256, 256, 0, stream>>>(context, ctx_bf);
    for (int d = 0; d < 2; ++d) {
        pack_b_kernel<<<(DC * AA + 255) / 256, 256, 0, stream>>>(dw[d].Wc, AA, Wc_p[d], DC, AA);
        pack_b_kernel<<<(HH * AA + 255) / 256, 256, 0, stream>>>(dw[d].Wh, AA, Wh_p[d], HH, AA);
        pack_b_kernel<<<(DIN * GG + 255) / 256, 256, 0, stream>>>(dw[d].Wih, GG, WihT_p[d], DIN, GG);
        pack_b_kernel<<<(DC * GG + 255) / 256, 256, 0, stream>>>(dw[d].Wih + (size_t)DIN * GG, GG, WihB_p[d], DC, GG);
        pack_b_kernel<<<(HH * GG + 255) / 256, 256, 0, stream>>>(dw[d].Whh, GG, Whh_p[d], HH, GG);
        zero_u32_kernel<<<(BB * HH) / 256, 256, 0, stream>>>((unsigned int*)hbuf[d]);
        zero_u32_kernel<<<(BB * HH / 2) / 256, 256, 0, stream>>>((unsigned int*)hbf[d]);
    }

    // ---- both directions ----
    for (int d = 0; d < 2; ++d) {
        // ctx_proj = context @ Wc  (M=B*L=16384, N=A, K=DC), bf16 output
        wmma_gemm<false, false, true><<<dim3(AA / 64, (BB * LL) / 64), 128, 0, stream>>>(
            ctx_bf, DC, Wc_p[d], nullptr, nullptr, ctxp, BB * LL, AA, DC);

        for (int s = 0; s < TT; ++s) {
            const int t = (d == 0) ? s : (TT - 1 - s);

            // hWh = h @ Wh   (64 x 1024 x 1024)
            wmma_gemm<false, false, false><<<dim3(AA / 64, 1), 128, 0, stream>>>(
                hbf[d], HH, Wh_p[d], nullptr, hWh, nullptr, BB, AA, HH);

            // attention -> c (bf16, B x DC)
            attn_kernel<<<BB, 256, 0, stream>>>(ctxp, hWh, dw[d].v, context, c_bf);

            // gi = x_t @ Wih_top + bih   (rows strided by T*DIN, base + t*DIN)
            wmma_gemm<true, false, false><<<dim3(GG / 64, 1), 128, 0, stream>>>(
                q_bf + (size_t)t * DIN, (long)TT * DIN, WihT_p[d], dw[d].bih,
                gi, nullptr, BB, GG, DIN);

            // gi += c @ Wih_bot
            wmma_gemm<false, true, false><<<dim3(GG / 64, 1), 128, 0, stream>>>(
                c_bf, DC, WihB_p[d], nullptr, gi, nullptr, BB, GG, DC);

            // gh = h @ Whh + bhh
            wmma_gemm<true, false, false><<<dim3(GG / 64, 1), 128, 0, stream>>>(
                hbf[d], HH, Whh_p[d], dw[d].bhh, gh, nullptr, BB, GG, HH);

            // GRU gates; writes h (f32+bf16) and output[t, :, dirOff:dirOff+H]
            gate_kernel<<<(BB * HH) / 256, 256, 0, stream>>>(
                gi, gh, hbuf[d], hbf[d], out, t, d * HH);
        }
    }

    hidden_kernel<<<(BB * HH) / 256, 256, 0, stream>>>(hbuf[0], hbuf[1], hidden);
}
